// GatingNetworkWithDecompWithTopK_84765474554321
// MI455X (gfx1250) — compile-verified, matrix operand order checked by reading, not run on hardware
//
#include <hip/hip_runtime.h>
#include <hip/hip_bf16.h>

// ---------------------------------------------------------------------------
// GatingNetworkWithDecompWithTopK — CDNA5 (gfx1250) bf16 WMMA implementation
// B=32, L=512, D=512, H=1024, E=8
// ---------------------------------------------------------------------------

#define B_  32
#define L_  512
#define D_  512
#define H_  1024
#define E_  8

typedef __attribute__((ext_vector_type(16))) __bf16 v16bf;
typedef __attribute__((ext_vector_type(8)))  float  v8f;

union FragU { v16bf v; uint4 u[2]; };

__device__ __forceinline__ unsigned short f2bf(float f) {
    unsigned u = __float_as_uint(f);
    unsigned r = (u + 0x7FFFu + ((u >> 16) & 1u)) >> 16;   // RNE
    return (unsigned short)r;
}

// --------------------------------------------------------------------------
// Kernel 1: repack Wt/Ws/W2 (fp32, row-major KxN) into per-lane WMMA-fragment
// order (bf16). Layout: [ktile][ntile][lane][16] with the 16x16x32 bf16 A/B
// lane pattern: n = nt*16 + (lane&15), k = ks*32 + (lane>>4)*8 + (i<8?i:16+i-8)
// Also zeroes the 32 per-batch mask-bit words (fresh every launch).
// --------------------------------------------------------------------------
__device__ __forceinline__ void pack_one(const float* __restrict__ W,
                                         unsigned short* __restrict__ Wp,
                                         int o, int nstride) {
    int i    = o & 15;
    int lane = (o >> 4) & 31;
    int tile = o >> 9;
    int nt   = tile & 63;     // 64 n-tiles (H=1024)
    int ks   = tile >> 6;     // 16 k-steps (D=512)
    int hl   = lane >> 4;
    int n    = nt * 16 + (lane & 15);
    int k    = ks * 32 + hl * 8 + (i < 8 ? i : 16 + (i - 8));
    Wp[o] = f2bf(W[k * nstride + n]);
}

__global__ __launch_bounds__(256) void repack_kernel(
        const float* __restrict__ Wt, const float* __restrict__ Ws,
        const float* __restrict__ W2,
        unsigned short* __restrict__ WtP, unsigned short* __restrict__ WsP,
        unsigned short* __restrict__ W2P, unsigned* __restrict__ maskbits) {
    int gid = blockIdx.x * 256 + threadIdx.x;
    if (gid < B_) maskbits[gid] = 0u;

    const int NBIG = (D_ / 32) * (H_ / 16) * 32 * 16;   // 524288 per matrix
    if (gid < NBIG) {
        pack_one(Wt, WtP, gid, H_);
    } else if (gid < 2 * NBIG) {
        pack_one(Ws, WsP, gid - NBIG, H_);
    } else if (gid < 2 * NBIG + (H_ / 32) * 32 * 16) {
        int o    = gid - 2 * NBIG;                      // [0, 16384)
        int i    = o & 15;
        int lane = (o >> 4) & 31;
        int ks   = o >> 9;                              // 0..31 (K=H=1024)
        int hl   = lane >> 4;
        int n    = lane & 15;                           // pad N=8..15 with 0
        int k    = ks * 32 + hl * 8 + (i < 8 ? i : 16 + (i - 8));
        float v  = (n < E_) ? W2[k * E_ + n] : 0.0f;
        W2P[o] = f2bf(v);
    }
}

// --------------------------------------------------------------------------
// Kernel 2: series decomposition. One thread per (b,d) column, rolling window
// sum over L with replicate-pad clamping. Emits bf16 trend & seasonal.
// --------------------------------------------------------------------------
__global__ __launch_bounds__(256) void decomp_kernel(
        const float* __restrict__ x,
        unsigned short* __restrict__ trendB,
        unsigned short* __restrict__ seasB) {
    int gid = blockIdx.x * 256 + threadIdx.x;   // 16384 threads
    int b = gid >> 9;
    int d = gid & (D_ - 1);
    const float* xb = x + (size_t)b * L_ * D_ + d;

    float s = 0.0f;
    #pragma unroll
    for (int j = -12; j <= 12; ++j) {
        int jj = j < 0 ? 0 : j;
        s += xb[(size_t)jj * D_];
    }
    for (int l = 0; l < L_; ++l) {
        float ma = s * (1.0f / 25.0f);
        float xv = xb[(size_t)l * D_];
        size_t o = ((size_t)b * L_ + l) * D_ + d;
        trendB[o] = f2bf(xv - ma);
        seasB[o]  = f2bf(ma);
        int add = l + 13; if (add > L_ - 1) add = L_ - 1;
        int sub = l - 12; if (sub < 0)      sub = 0;
        s += xb[(size_t)add * D_] - xb[(size_t)sub * D_];
    }
}

// --------------------------------------------------------------------------
// Kernel 3: fused dual GEMM (trend@Wt, seasonal@Ws) + bias + LayerNorm x2 +
// relu(t+s) + GEMM2(@W2) + softmax + top-2 mask bits + gate-score rows.
// Block = 16 rows x full H. 8 waves, each owns 128 columns.
// --------------------------------------------------------------------------
__global__ __launch_bounds__(256) void fused_kernel(
        const unsigned short* __restrict__ trendB,
        const unsigned short* __restrict__ seasB,
        const unsigned short* __restrict__ WtP,
        const unsigned short* __restrict__ WsP,
        const unsigned short* __restrict__ W2P,
        const float* __restrict__ bt,  const float* __restrict__ gt,
        const float* __restrict__ bet, const float* __restrict__ bs,
        const float* __restrict__ gsg, const float* __restrict__ bes,
        const float* __restrict__ b2,
        float* __restrict__ gsbuf, unsigned* __restrict__ maskbits) {
    __shared__ float redT[16], redT2[16], redS[16], redS2[16];
    __shared__ unsigned short medS[16 * H_];   // 32 KB
    __shared__ float logitsS[16 * 16];

    const int tid  = threadIdx.x;
    const int w    = tid >> 5;
    const int lane = tid & 31;
    const int hl   = lane >> 4;
    const int lm   = lane & 15;
    const int m0   = blockIdx.x * 16;

    if (tid < 16) { redT[tid] = 0.f; redT2[tid] = 0.f; redS[tid] = 0.f; redS2[tid] = 0.f; }
    __syncthreads();

    const v8f vzero = {0.f, 0.f, 0.f, 0.f, 0.f, 0.f, 0.f, 0.f};
    v8f tAcc[8], sAcc[8];
    #pragma unroll
    for (int nt = 0; nt < 8; ++nt) { tAcc[nt] = vzero; sAcc[nt] = vzero; }

    const unsigned aRow = (unsigned)(m0 + lm) * D_;

    for (int ks = 0; ks < D_ / 32; ++ks) {
        const int kb = ks * 32 + hl * 8;
        FragU at, as_;
        at.u[0]  = *(const uint4*)(trendB + aRow + kb);
        at.u[1]  = *(const uint4*)(trendB + aRow + kb + 16);
        as_.u[0] = *(const uint4*)(seasB + aRow + kb);
        as_.u[1] = *(const uint4*)(seasB + aRow + kb + 16);

        if (ks + 1 < D_ / 32) {  // prefetch next K-step's B tiles (gfx1250 global_prefetch)
            __builtin_prefetch(WtP + ((unsigned)((ks + 1) * 64 + w * 8) * 32u + lane) * 16u, 0, 1);
            __builtin_prefetch(WsP + ((unsigned)((ks + 1) * 64 + w * 8) * 32u + lane) * 16u, 0, 1);
        }

        #pragma unroll
        for (int nt = 0; nt < 8; ++nt) {
            const unsigned tileIdx = (unsigned)(ks * 64 + w * 8 + nt);
            const uint4* btp = (const uint4*)(WtP + (tileIdx * 32u + lane) * 16u);
            FragU bft; bft.u[0] = btp[0]; bft.u[1] = btp[1];
            tAcc[nt] = __builtin_amdgcn_wmma_f32_16x16x32_bf16(
                false, at.v, false, bft.v, (short)0, tAcc[nt], false, false);
            const uint4* bsp = (const uint4*)(WsP + (tileIdx * 32u + lane) * 16u);
            FragU bfs; bfs.u[0] = bsp[0]; bfs.u[1] = bsp[1];
            sAcc[nt] = __builtin_amdgcn_wmma_f32_16x16x32_bf16(
                false, as_.v, false, bfs.v, (short)0, sAcc[nt], false, false);
        }
    }

    // bias + per-row partial sums / sums-of-squares
    float sumT[8], sqT[8], sumS[8], sqS[8];
    #pragma unroll
    for (int r = 0; r < 8; ++r) { sumT[r] = 0.f; sqT[r] = 0.f; sumS[r] = 0.f; sqS[r] = 0.f; }
    #pragma unroll
    for (int nt = 0; nt < 8; ++nt) {
        const int n = w * 128 + nt * 16 + lm;
        const float btv = bt[n], bsv = bs[n];
        #pragma unroll
        for (int r = 0; r < 8; ++r) {
            float tv = tAcc[nt][r] + btv; tAcc[nt][r] = tv;
            sumT[r] += tv; sqT[r] += tv * tv;
            float sv = sAcc[nt][r] + bsv; sAcc[nt][r] = sv;
            sumS[r] += sv; sqS[r] += sv * sv;
        }
    }
    #pragma unroll
    for (int r = 0; r < 8; ++r) {
        const int m = r + hl * 8;
        atomicAdd(&redT[m],  sumT[r]);
        atomicAdd(&redT2[m], sqT[r]);
        atomicAdd(&redS[m],  sumS[r]);
        atomicAdd(&redS2[m], sqS[r]);
    }
    __syncthreads();

    // LayerNorm params per row
    float muT[8], rsT[8], muS[8], rsS[8];
    #pragma unroll
    for (int r = 0; r < 8; ++r) {
        const int m = r + hl * 8;
        float mt = redT[m] * (1.0f / H_);
        float vt = redT2[m] * (1.0f / H_) - mt * mt;
        muT[r] = mt; rsT[r] = rsqrtf(vt + 1e-5f);
        float ms = redS[m] * (1.0f / H_);
        float vs = redS2[m] * (1.0f / H_) - ms * ms;
        muS[r] = ms; rsS[r] = rsqrtf(vs + 1e-5f);
    }
    // LN + relu(t+s) -> med (bf16, LDS)
    #pragma unroll
    for (int nt = 0; nt < 8; ++nt) {
        const int n = w * 128 + nt * 16 + lm;
        const float gtv = gt[n], betv = bet[n], gsv = gsg[n], besv = bes[n];
        #pragma unroll
        for (int r = 0; r < 8; ++r) {
            const int m = r + hl * 8;
            float tv = (tAcc[nt][r] - muT[r]) * rsT[r] * gtv + betv;
            float sv = (sAcc[nt][r] - muS[r]) * rsS[r] * gsv + besv;
            float md = tv + sv;
            md = md > 0.f ? md : 0.f;
            medS[m * H_ + n] = f2bf(md);
        }
    }
    __syncthreads();

    // GEMM2: med[16x1024] @ W2p[1024x16] (cols >= 8 are zero-padded)
    if (w == 0) {
        v8f acc2 = vzero;
        for (int ks = 0; ks < H_ / 32; ++ks) {
            const int kb = ks * 32 + hl * 8;
            FragU am;
            am.u[0] = *(const uint4*)(&medS[lm * H_ + kb]);
            am.u[1] = *(const uint4*)(&medS[lm * H_ + kb + 16]);
            const uint4* bp = (const uint4*)(W2P + ((unsigned)(ks * 32 + lane)) * 16u);
            FragU bf2; bf2.u[0] = bp[0]; bf2.u[1] = bp[1];
            acc2 = __builtin_amdgcn_wmma_f32_16x16x32_bf16(
                false, am.v, false, bf2.v, (short)0, acc2, false, false);
        }
        const float b2v = (lm < E_) ? b2[lm] : 0.0f;
        #pragma unroll
        for (int r = 0; r < 8; ++r)
            logitsS[(r + hl * 8) * 16 + lm] = acc2[r] + b2v;
    }
    __syncthreads();

    // softmax over E=8, top-2 -> mask bits, store gate rows for l < 8
    if (tid < 16) {
        const int g = m0 + tid;
        const int b = g >> 9;          // / L
        const int l = g & (L_ - 1);    // % L
        float v[E_];
        float mx = -3.4e38f;
        #pragma unroll
        for (int e = 0; e < E_; ++e) { v[e] = logitsS[tid * 16 + e]; mx = fmaxf(mx, v[e]); }
        float sum = 0.f;
        #pragma unroll
        for (int e = 0; e < E_; ++e) { v[e] = __expf(v[e] - mx); sum += v[e]; }
        const float inv = 1.0f / sum;
        int j0 = 0;
        #pragma unroll
        for (int e = 1; e < E_; ++e) if (v[e] > v[j0]) j0 = e;
        int j1 = -1;
        #pragma unroll
        for (int e = 0; e < E_; ++e)
            if (e != j0 && (j1 < 0 || v[e] > v[j1])) j1 = e;
        atomicOr(&maskbits[b], (1u << (j0 * 2)) | (1u << (j1 * 2 + 1)));
        if (l < E_) {
            #pragma unroll
            for (int e = 0; e < E_; ++e)
                gsbuf[((b * E_ + l) * E_) + e] = v[e] * inv;
        }
    }
}

// --------------------------------------------------------------------------
// Kernel 4: zero output; for the 16 live (l,e) pairs compute cross-batch
// denominator and write  masked/denom * capacity(=64).
// --------------------------------------------------------------------------
__global__ __launch_bounds__(256) void finalize_kernel(
        const float* __restrict__ gsbuf, const unsigned* __restrict__ maskbits,
        float* __restrict__ out) {
    const int total = B_ * L_ * E_;
    for (int i = blockIdx.x * blockDim.x + threadIdx.x; i < total;
         i += gridDim.x * blockDim.x) {
        int e = i & (E_ - 1);
        int l = (i >> 3) & (L_ - 1);
        if (!(l < E_ && e < 2)) out[i] = 0.0f;
    }
    if (blockIdx.x == 0 && threadIdx.x < 16) {
        const int l = threadIdx.x >> 1;
        const int e = threadIdx.x & 1;
        float denom = 1e-4f;
        float vals[B_];
        #pragma unroll
        for (int b = 0; b < B_; ++b) {
            float mk = ((maskbits[b] >> (l * 2 + e)) & 1u) ? 1.0f : 0.0f;
            float vv = gsbuf[(b * E_ + l) * E_ + e] * mk;
            vals[b] = vv;
            denom += vv;
        }
        const float sc = 64.0f / denom;   // capacity = int(2.0 * 32)
        #pragma unroll
        for (int b = 0; b < B_; ++b)
            out[((size_t)b * L_ + l) * E_ + e] = vals[b] * sc;
    }
}

// --------------------------------------------------------------------------
extern "C" void kernel_launch(void* const* d_in, const int* in_sizes, int n_in,
                              void* d_out, int out_size, void* d_ws, size_t ws_size,
                              hipStream_t stream) {
    const float* x   = (const float*)d_in[0];
    const float* Wt  = (const float*)d_in[1];
    const float* bt  = (const float*)d_in[2];
    const float* gt  = (const float*)d_in[3];
    const float* bet = (const float*)d_in[4];
    const float* Ws  = (const float*)d_in[5];
    const float* bs  = (const float*)d_in[6];
    const float* gsg = (const float*)d_in[7];
    const float* bes = (const float*)d_in[8];
    const float* W2  = (const float*)d_in[9];
    const float* b2  = (const float*)d_in[10];

    char* ws = (char*)d_ws;
    const size_t SZ_ACT  = (size_t)B_ * L_ * D_ * 2;            // 16 MB bf16
    const size_t SZ_WBIG = (size_t)D_ * H_ * 2;                 // 1 MB bf16
    const size_t SZ_W2P  = (size_t)(H_ / 32) * 32 * 16 * 2;     // 32 KB
    const size_t SZ_GS   = (size_t)B_ * E_ * E_ * 4;            // 8 KB

    unsigned short* trendB = (unsigned short*)(ws);
    unsigned short* seasB  = (unsigned short*)(ws + SZ_ACT);
    unsigned short* WtP    = (unsigned short*)(ws + 2 * SZ_ACT);
    unsigned short* WsP    = (unsigned short*)(ws + 2 * SZ_ACT + SZ_WBIG);
    unsigned short* W2P    = (unsigned short*)(ws + 2 * SZ_ACT + 2 * SZ_WBIG);
    float*          gsbuf  = (float*)(ws + 2 * SZ_ACT + 2 * SZ_WBIG + SZ_W2P);
    unsigned*       maskb  = (unsigned*)(ws + 2 * SZ_ACT + 2 * SZ_WBIG + SZ_W2P + SZ_GS);

    // 1) pack weights to bf16 fragment layout (+ zero mask bits)
    const int repack_total = 2 * 524288 + 16384;                // 1064960
    repack_kernel<<<(repack_total + 255) / 256, 256, 0, stream>>>(
        Wt, Ws, W2, WtP, WsP, W2P, maskb);

    // 2) moving-average decomposition -> bf16 trend/seasonal
    decomp_kernel<<<(B_ * D_) / 256, 256, 0, stream>>>(x, trendB, seasB);

    // 3) fused dual-GEMM + LN + relu + GEMM2 + softmax + top-2
    fused_kernel<<<(B_ * L_) / 16, 256, 0, stream>>>(
        trendB, seasB, WtP, WsP, W2P, bt, gt, bet, bs, gsg, bes, b2,
        gsbuf, maskb);

    // 4) zero output + gated normalization
    finalize_kernel<<<128, 256, 0, stream>>>(gsbuf, maskb, (float*)d_out);
}
